// RNNModel_52063593562843
// MI455X (gfx1250) — compile-verified, matrix operand order checked by reading, not run on hardware
//
#include <hip/hip_runtime.h>
#include <hip/hip_bf16.h>

// ---------------------------------------------------------------------------
// Types for CDNA5 WMMA (wave32) and TDM descriptors.
// ---------------------------------------------------------------------------
typedef __bf16 bf16;
typedef __attribute__((ext_vector_type(16))) __bf16 v16bf;
typedef __attribute__((ext_vector_type(8)))  __bf16 v8bf;
typedef __attribute__((ext_vector_type(4)))  __bf16 v4bf;
typedef __attribute__((ext_vector_type(8)))  float  v8f;
typedef unsigned int u32;
typedef unsigned long long u64;
typedef __attribute__((ext_vector_type(4))) u32 u32x4;
typedef __attribute__((ext_vector_type(8))) int i32x8;
typedef __attribute__((ext_vector_type(4))) int i32x4;

#define HAS_TDM (__has_builtin(__builtin_amdgcn_tensor_load_to_lds) && \
                 __has_builtin(__builtin_amdgcn_s_wait_tensorcnt))

__device__ __forceinline__ v16bf cat16(v8bf lo, v8bf hi) {
    return __builtin_shufflevector(lo, hi, 0,1,2,3,4,5,6,7,8,9,10,11,12,13,14,15);
}

__device__ __forceinline__ v8f wmma_bf16(v16bf a, v16bf b, v8f c) {
    return __builtin_amdgcn_wmma_f32_16x16x32_bf16(false, a, false, b, (short)0, c, false, false);
}

__device__ __forceinline__ float sigmoidf(float x) { return 1.0f / (1.0f + __expf(-x)); }

// Low 32 bits of a generic pointer to an LDS object = LDS byte offset
// (flat-LDS aperture: the LDS address is the low 32 address bits, ISA 10.2).
__device__ __forceinline__ u32 lds_off_of(const void* p) {
    return (u32)(u64)(uintptr_t)p;
}

#if HAS_TDM
// Issue one TDM 2D tile load: dim1 rows x dim0 elements of 2 bytes,
// row stride stride0 elements, global -> LDS. D# per CDNA5 ISA ch.8.
// clang-23 / amdgpu-toolchain lane: 6-arg builtin.
__device__ __forceinline__ void tdm_load_2d_bf16(u32 lds_off, const void* gptr,
                                                 u32 dim0, u32 dim1, u32 stride0) {
    u64 ga = (u64)(uintptr_t)gptr;
    u64 s1 = (u64)stride0 * dim1;           // dim1 stride (unused for 2D tiles)
    u32x4 g0;
    g0[0] = 1u;                                              // count=1, user D#
    g0[1] = lds_off;                                         // lds_addr
    g0[2] = (u32)ga;                                         // global_addr[31:0]
    g0[3] = (u32)((ga >> 32) & 0x1FFFFFFu) | (2u << 30);     // global_addr[56:32] | type=2
    i32x8 g1;
    g1[0] = (int)(1u << 16);                                 // wg_mask=0, data_size=1 (2B)
    g1[1] = (int)((dim0 & 0xFFFFu) << 16);                   // tensor_dim0[15:0]
    g1[2] = (int)((dim0 >> 16) | ((dim1 & 0xFFFFu) << 16));  // dim0[31:16] | dim1[15:0]
    g1[3] = (int)((dim1 >> 16) | ((dim0 & 0xFFFFu) << 16));  // dim1[31:16] | tile_dim0
    g1[4] = (int)(dim1 & 0xFFFFu);                           // tile_dim1 | tile_dim2=0
    g1[5] = (int)(stride0);                                  // tensor_dim0_stride[31:0]
    g1[6] = (int)(((u32)s1 & 0xFFFFu) << 16);                // s0[47:32]=0 | s1[15:0]
    g1[7] = (int)((u32)(s1 >> 16));                          // s1[47:16]
    i32x4 z4 = {0, 0, 0, 0};
    i32x8 z8 = {0, 0, 0, 0, 0, 0, 0, 0};
    __builtin_amdgcn_tensor_load_to_lds(g0, g1, z4, z4, z8, 0);
}
#endif

// ---------------------------------------------------------------------------
// Utility kernels
// ---------------------------------------------------------------------------
__global__ void f32_to_bf16_kernel(const float* __restrict__ in, bf16* __restrict__ out, int n) {
    int i = blockIdx.x * blockDim.x + threadIdx.x;
    if (i < n) out[i] = (bf16)in[i];
}

__global__ void zero_ints_kernel(int* p, int n) {
    int i = threadIdx.x;
    if (i < n) p[i] = 0;
}

// ---------------------------------------------------------------------------
// Generic WMMA GEMM:  C[M x N] = A[M x K](lda) * W[N x K]^T + b0 + b1
// BM=BN=BK=64; 256 threads = 8 waves; each wave owns two 16x16 output tiles
// sharing one B fragment. Register-staged LDS double buffering: global loads
// for tile k+1 are in flight while tile k is computed (no serial loadcnt
// stalls), staging is branchless via clamped row indices.
// ---------------------------------------------------------------------------
#define BM 64
#define BN 64
#define BK 64

__global__ void __launch_bounds__(256)
gemm_bias_kernel(const float* __restrict__ A, const float* __restrict__ W,
                 const float* __restrict__ b0, const float* __restrict__ b1,
                 float* __restrict__ C, int M, int N, int K, long lda)
{
    __shared__ bf16 As[2][BM][BK];    // 16 KB
    __shared__ bf16 Bs[2][BN][BK];    // 16 KB

    const int bm   = blockIdx.y * BM;
    const int bn   = blockIdx.x * BN;
    const int tid  = threadIdx.x;
    const int lane = tid & 31;
    const int wave = tid >> 5;          // 0..7
    const int t0   = wave * 2;
    const int tm0  = t0 & 3;
    const int tm1  = (t0 + 1) & 3;
    const int tn   = t0 >> 2;

    v8f acc0 = {}; v8f acc1 = {};

    const int m  = lane & 15;
    const int hh = lane >> 4;

    // Per-thread staging coordinates: 4 chunks of float4 per matrix.
    int rr[4], cc4[4], gmr[4], gnr[4];
    const float* pa[4]; const float* pw[4];
    #pragma unroll
    for (int u = 0; u < 4; ++u) {
        const int i = tid + u * 256;
        rr[u]  = i >> 4;
        cc4[u] = (i & 15) * 4;
        int gm = bm + rr[u];
        int gn = bn + rr[u];
        gmr[u] = gm;
        gnr[u] = gn;
        gm = (gm < M) ? gm : (M - 1);       // clamp: load always in-bounds
        gn = (gn < N) ? gn : (N - 1);
        pa[u] = A + (size_t)gm * lda + cc4[u];
        pw[u] = W + (size_t)gn * K + cc4[u];
    }

    float4 va[4], vb[4];
    // Prologue: load k-tile 0.
    #pragma unroll
    for (int u = 0; u < 4; ++u) { va[u] = *(const float4*)pa[u]; vb[u] = *(const float4*)pw[u]; }
    #pragma unroll
    for (int u = 0; u < 4; ++u) {
        float4 a = (gmr[u] < M) ? va[u] : make_float4(0.f, 0.f, 0.f, 0.f);
        float4 b = (gnr[u] < N) ? vb[u] : make_float4(0.f, 0.f, 0.f, 0.f);
        *(v4bf*)&As[0][rr[u]][cc4[u]] = (v4bf){ (bf16)a.x, (bf16)a.y, (bf16)a.z, (bf16)a.w };
        *(v4bf*)&Bs[0][rr[u]][cc4[u]] = (v4bf){ (bf16)b.x, (bf16)b.y, (bf16)b.z, (bf16)b.w };
    }
    __syncthreads();

    int cur = 0;
    for (int k0 = 0; k0 < K; k0 += BK) {
        const bool more = (k0 + BK) < K;
        if (more) {     // issue next tile's loads; latency hidden under WMMA
            #pragma unroll
            for (int u = 0; u < 4; ++u) {
                va[u] = *(const float4*)(pa[u] + k0 + BK);
                vb[u] = *(const float4*)(pw[u] + k0 + BK);
            }
        }

        #pragma unroll
        for (int kk = 0; kk < BK; kk += 32) {
            const bf16* pb = &Bs[cur][tn * 16 + m][kk + hh * 8];
            v16bf bfr = cat16(*(const v8bf*)pb, *(const v8bf*)(pb + 16));

            const bf16* pa0 = &As[cur][tm0 * 16 + m][kk + hh * 8];
            acc0 = wmma_bf16(cat16(*(const v8bf*)pa0, *(const v8bf*)(pa0 + 16)), bfr, acc0);

            const bf16* pa1 = &As[cur][tm1 * 16 + m][kk + hh * 8];
            acc1 = wmma_bf16(cat16(*(const v8bf*)pa1, *(const v8bf*)(pa1 + 16)), bfr, acc1);
        }

        if (more) {
            const int nxt = cur ^ 1;
            #pragma unroll
            for (int u = 0; u < 4; ++u) {
                float4 a = (gmr[u] < M) ? va[u] : make_float4(0.f, 0.f, 0.f, 0.f);
                float4 b = (gnr[u] < N) ? vb[u] : make_float4(0.f, 0.f, 0.f, 0.f);
                *(v4bf*)&As[nxt][rr[u]][cc4[u]] = (v4bf){ (bf16)a.x, (bf16)a.y, (bf16)a.z, (bf16)a.w };
                *(v4bf*)&Bs[nxt][rr[u]][cc4[u]] = (v4bf){ (bf16)b.x, (bf16)b.y, (bf16)b.z, (bf16)b.w };
            }
            __syncthreads();
            cur = nxt;
        }
    }

    // Epilogue: D row = r + 8*(lane>>4), col = lane&15
    const int col   = lane & 15;
    const int rbase = hh * 8;
    const int gn    = bn + tn * 16 + col;
    if (gn < N) {
        float bias = (b0 ? b0[gn] : 0.0f) + (b1 ? b1[gn] : 0.0f);
        #pragma unroll
        for (int r = 0; r < 8; ++r) {
            int gm0 = bm + tm0 * 16 + rbase + r;
            if (gm0 < M) C[(size_t)gm0 * N + gn] = acc0[r] + bias;
            int gm1 = bm + tm1 * 16 + rbase + r;
            if (gm1 < M) C[(size_t)gm1 * N + gn] = acc1[r] + bias;
        }
    }
}

// ---------------------------------------------------------------------------
// Persistent recurrent scan.
//   16 WGs x 512 threads. WG j owns h columns [32j, 32j+32).
//   Whh slice (NGATES*32 rows x 512, bf16) resident in LDS for all T steps,
//   preloaded via TDM. Per step: TDM-broadcast h (32x512 bf16) into LDS,
//   16 WMMAs per active wave with LDS-only operands, elementwise cell update,
//   global barrier. h state is double-buffered bf16 in global ws.
// ---------------------------------------------------------------------------
#define NWG_SCAN 16
#define BATCH 32
#define HDIM  512

__device__ __forceinline__ void grid_barrier(int* cnt, int target) {
    __threadfence();
    __syncthreads();
    if (threadIdx.x == 0) {
        atomicAdd(cnt, 1);
        volatile int* vc = cnt;
        while (*vc < target) { __builtin_amdgcn_s_sleep(1); }
    }
    __syncthreads();
    __threadfence();
}

template <int NGATES, int CELL>   // CELL: 0=LSTM, 1=GRU, 2=RNN-tanh
__global__ void __launch_bounds__(512, 1)
scan_kernel(const float* __restrict__ xg,   // B x T x (NGATES*HDIM)
            const bf16*  __restrict__ whh,  // (NGATES*HDIM) x HDIM bf16
            const float* __restrict__ bhh,  // GRU recurrent bias (else unused)
            bf16*  __restrict__ hbuf,       // 2 x B x HDIM bf16 (double buffer)
            float* __restrict__ cbuf,       // B x HDIM (LSTM only)
            float* __restrict__ out,        // B x T x HDIM f32
            int* __restrict__ bar, int T)
{
    __shared__ bf16  Hs[BATCH][HDIM];              // 32 KB
    __shared__ bf16  Ws[NGATES * 32][HDIM];        // 32..128 KB (LDS-resident weights)
    __shared__ float Gs[4][BATCH][32];             // 16 KB (max 4 gates)

    const int j    = blockIdx.x;
    const int hc   = j * 32;
    const int tid  = threadIdx.x;
    const int lane = tid & 31;
    const int wave = tid >> 5;
    const int m    = lane & 15;
    const int hh   = lane >> 4;

    int nbar = 0;

    // --- One-time preload of this WG's Whh slice into LDS ---
#if HAS_TDM
    if (tid == 0) {
        #pragma unroll
        for (int g = 0; g < NGATES; ++g) {
            tdm_load_2d_bf16(lds_off_of(&Ws[g * 32][0]),
                             whh + ((size_t)g * HDIM + hc) * HDIM,
                             HDIM, 32, HDIM);
        }
        __builtin_amdgcn_s_wait_tensorcnt(0);
    }
#else
    for (int i = tid; i < NGATES * 32 * HDIM / 8; i += 512) {
        int g = i / (32 * HDIM / 8);
        int rem = i - g * (32 * HDIM / 8);
        int r = rem >> 6, c = (rem & 63) * 8;           // 8 bf16 = 16B chunks
        *(v8bf*)&Ws[g * 32 + r][c] =
            *(const v8bf*)(whh + ((size_t)(g * HDIM + hc + r)) * HDIM + c);
    }
#endif

    // init owned h slice (parity 0) and c
    for (int i = tid; i < BATCH * 32; i += 512) {
        int b = i >> 5, cc = i & 31;
        hbuf[b * HDIM + hc + cc] = (bf16)0.0f;
        if (CELL == 0) cbuf[b * HDIM + hc + cc] = 0.0f;
    }
    grid_barrier(bar, (++nbar) * NWG_SCAN);

    for (int t = 0; t < T; ++t) {
        const bf16* hcur = hbuf + (size_t)(t & 1) * (BATCH * HDIM);
        bf16*       hnxt = hbuf + (size_t)((t + 1) & 1) * (BATCH * HDIM);

        // --- broadcast full h (32x512 bf16) into LDS ---
#if HAS_TDM
        if (tid == 0) {
            tdm_load_2d_bf16(lds_off_of(&Hs[0][0]), hcur, HDIM, BATCH, HDIM);
            __builtin_amdgcn_s_wait_tensorcnt(0);
        }
#else
        for (int i = tid; i < BATCH * HDIM / 8; i += 512) {
            int b = i >> 6, c = (i & 63) * 8;
            *(v8bf*)&Hs[b][c] = *(const v8bf*)(hcur + b * HDIM + c);
        }
#endif
        __syncthreads();

        // --- recurrent matmul: one 16x16 tile per active wave, LDS operands ---
        if (wave < 4 * NGATES) {
            const int gate = wave >> 2;
            const int sub  = wave & 3;
            const int tm   = sub & 1;
            const int tn   = sub >> 1;
            const bf16* wrow = &Ws[gate * 32 + tn * 16 + m][0];

            v8f acc = {};
            #pragma unroll 4
            for (int k0 = 0; k0 < HDIM; k0 += 32) {
                const bf16* pa = &Hs[tm * 16 + m][k0 + hh * 8];
                v16bf af = cat16(*(const v8bf*)pa, *(const v8bf*)(pa + 16));
                const bf16* pb = wrow + k0 + hh * 8;
                v16bf bfr = cat16(*(const v8bf*)pb, *(const v8bf*)(pb + 16));
                acc = wmma_bf16(af, bfr, acc);
            }
            const int col = lane & 15;
            #pragma unroll
            for (int r = 0; r < 8; ++r) {
                Gs[gate][tm * 16 + hh * 8 + r][tn * 16 + col] = acc[r];
            }
        }
        __syncthreads();

        // --- elementwise cell update on owned 32x32 slice ---
        for (int i = tid; i < BATCH * 32; i += 512) {
            int b = i >> 5, cc = i & 31;
            int gcol = hc + cc;
            const float* xgrow = xg + ((size_t)b * T + t) * (NGATES * HDIM);
            float h;
            if (CELL == 0) {           // LSTM (xg has b_ih+b_hh), gates i,f,g,o
                float gi = sigmoidf(Gs[0][b][cc] + xgrow[0 * HDIM + gcol]);
                float gf = sigmoidf(Gs[1][b][cc] + xgrow[1 * HDIM + gcol]);
                float gg = tanhf   (Gs[2][b][cc] + xgrow[2 * HDIM + gcol]);
                float go = sigmoidf(Gs[3][b][cc] + xgrow[3 * HDIM + gcol]);
                float c  = gf * cbuf[b * HDIM + gcol] + gi * gg;
                h = go * tanhf(c);
                cbuf[b * HDIM + gcol] = c;
            } else if (CELL == 1) {    // GRU (xg has b_ih), gates r,z,n
                float r  = sigmoidf(xgrow[0 * HDIM + gcol] + Gs[0][b][cc] + bhh[0 * HDIM + gcol]);
                float z  = sigmoidf(xgrow[1 * HDIM + gcol] + Gs[1][b][cc] + bhh[1 * HDIM + gcol]);
                float n  = tanhf   (xgrow[2 * HDIM + gcol] +
                                    r * (Gs[2][b][cc] + bhh[2 * HDIM + gcol]));
                h = (1.0f - z) * n + z * (float)hcur[b * HDIM + gcol];
            } else {                   // RNN tanh (xg has b_ih+b_hh)
                h = tanhf(Gs[0][b][cc] + xgrow[gcol]);
            }
            hnxt[b * HDIM + gcol] = (bf16)h;
            out[((size_t)b * T + t) * HDIM + gcol] = h;
        }
        grid_barrier(bar, (++nbar) * NWG_SCAN);
    }
}

// ---------------------------------------------------------------------------
// LayerNorm (D=512), one 256-thread block per row, wave32 shuffle reductions.
// ---------------------------------------------------------------------------
__global__ void __launch_bounds__(256)
layernorm_kernel(float* __restrict__ x, const float* __restrict__ g,
                 const float* __restrict__ b, int D)
{
    float* p = x + (size_t)blockIdx.x * D;
    __shared__ float rs[8], rs2[8];

    float s = 0.0f, s2 = 0.0f;
    for (int i = threadIdx.x; i < D; i += 256) {
        float v = p[i]; s += v; s2 += v * v;
    }
    #pragma unroll
    for (int o = 16; o; o >>= 1) { s += __shfl_xor(s, o, 32); s2 += __shfl_xor(s2, o, 32); }
    int lane = threadIdx.x & 31, wave = threadIdx.x >> 5;
    if (lane == 0) { rs[wave] = s; rs2[wave] = s2; }
    __syncthreads();
    if (wave == 0) {
        s  = (lane < 8) ? rs[lane]  : 0.0f;
        s2 = (lane < 8) ? rs2[lane] : 0.0f;
        #pragma unroll
        for (int o = 4; o; o >>= 1) { s += __shfl_xor(s, o, 32); s2 += __shfl_xor(s2, o, 32); }
        if (lane == 0) { rs[0] = s; rs2[0] = s2; }
    }
    __syncthreads();
    float mean = rs[0] / (float)D;
    float var  = rs2[0] / (float)D - mean * mean;
    float rstd = rsqrtf(var + 1e-5f);
    for (int i = threadIdx.x; i < D; i += 256) {
        p[i] = (p[i] - mean) * rstd * g[i] + b[i];
    }
}

// ---------------------------------------------------------------------------
// Host-side orchestration
// ---------------------------------------------------------------------------
extern "C" void kernel_launch(void* const* d_in, const int* in_sizes, int n_in,
                              void* d_out, int out_size, void* d_ws, size_t ws_size,
                              hipStream_t stream) {
    const int B = 32, T = 1024, I = 256, H = 512, O = 1000;
    const int M = B * T;

    const float* x         = (const float*)d_in[0];
    const float* lstm_w_ih = (const float*)d_in[1];
    const float* lstm_w_hh = (const float*)d_in[2];
    const float* lstm_b_ih = (const float*)d_in[3];
    const float* lstm_b_hh = (const float*)d_in[4];
    const float* ln1_g     = (const float*)d_in[5];
    const float* ln1_b     = (const float*)d_in[6];
    const float* gru_w_ih  = (const float*)d_in[7];
    const float* gru_w_hh  = (const float*)d_in[8];
    const float* gru_b_ih  = (const float*)d_in[9];
    const float* gru_b_hh  = (const float*)d_in[10];
    const float* ln2_g     = (const float*)d_in[11];
    const float* ln2_b     = (const float*)d_in[12];
    const float* rnn_w_ih  = (const float*)d_in[13];
    const float* rnn_w_hh  = (const float*)d_in[14];
    const float* rnn_b_ih  = (const float*)d_in[15];
    const float* rnn_b_hh  = (const float*)d_in[16];
    const float* fc_w      = (const float*)d_in[17];
    const float* fc_b      = (const float*)d_in[18];
    float* out             = (float*)d_out;

    char* w = (char*)d_ws;
    auto carve = [&](size_t bytes) -> void* {
        void* p = (void*)w;
        w += (bytes + 255) & ~(size_t)255;
        return p;
    };
    float* xg     = (float*)carve((size_t)M * (4 * H) * 4);   // 256 MB, reused
    float* hs1    = (float*)carve((size_t)M * H * 4);         // 64 MB
    float* hs2    = (float*)carve((size_t)M * H * 4);         // 64 MB
    bf16*  whh_l  = (bf16*)carve((size_t)4 * H * H * 2);
    bf16*  whh_g  = (bf16*)carve((size_t)3 * H * H * 2);
    bf16*  whh_r  = (bf16*)carve((size_t)H * H * 2);
    bf16*  hbuf_l = (bf16*)carve((size_t)2 * B * H * 2);
    float* cbuf_l = (float*)carve((size_t)B * H * 4);
    bf16*  hbuf_g = (bf16*)carve((size_t)2 * B * H * 2);
    bf16*  hbuf_r = (bf16*)carve((size_t)2 * B * H * 2);
    int*   bars   = (int*)carve(256);

    zero_ints_kernel<<<1, 32, 0, stream>>>(bars, 8);
    {
        int n = 4 * H * H;
        f32_to_bf16_kernel<<<(n + 255) / 256, 256, 0, stream>>>(lstm_w_hh, whh_l, n);
        n = 3 * H * H;
        f32_to_bf16_kernel<<<(n + 255) / 256, 256, 0, stream>>>(gru_w_hh, whh_g, n);
        n = H * H;
        f32_to_bf16_kernel<<<(n + 255) / 256, 256, 0, stream>>>(rnn_w_hh, whh_r, n);
    }

    // --- LSTM ---
    gemm_bias_kernel<<<dim3((4 * H) / BN, M / BM), 256, 0, stream>>>(
        x, lstm_w_ih, lstm_b_ih, lstm_b_hh, xg, M, 4 * H, I, I);
    scan_kernel<4, 0><<<NWG_SCAN, 512, 0, stream>>>(
        xg, whh_l, nullptr, hbuf_l, cbuf_l, hs1, bars + 0, T);
    layernorm_kernel<<<M, 256, 0, stream>>>(hs1, ln1_g, ln1_b, H);

    // --- GRU ---
    gemm_bias_kernel<<<dim3((3 * H) / BN, M / BM), 256, 0, stream>>>(
        hs1, gru_w_ih, gru_b_ih, nullptr, xg, M, 3 * H, H, H);
    scan_kernel<3, 1><<<NWG_SCAN, 512, 0, stream>>>(
        xg, whh_g, gru_b_hh, hbuf_g, nullptr, hs2, bars + 1, T);
    layernorm_kernel<<<M, 256, 0, stream>>>(hs2, ln2_g, ln2_b, H);

    // --- RNN tanh ---
    gemm_bias_kernel<<<dim3(H / BN, M / BM), 256, 0, stream>>>(
        hs2, rnn_w_ih, rnn_b_ih, rnn_b_hh, xg, M, H, H, H);
    scan_kernel<1, 2><<<NWG_SCAN, 512, 0, stream>>>(
        xg, whh_r, nullptr, hbuf_r, nullptr, hs1, bars + 2, T);

    // --- FC on last timestep (strided rows, lda = T*H) ---
    gemm_bias_kernel<<<dim3((O + BN - 1) / BN, 1), 256, 0, stream>>>(
        hs1 + (size_t)(T - 1) * H, fc_w, fc_b, nullptr, out, B, O, H, (long)T * H);
}
